// SSLPretrainModel_60876866453593
// MI455X (gfx1250) — matrix operand order repaired and checked by reading, not compiled, not run on hardware
//
#include <hip/hip_runtime.h>
#include <hip/hip_bf16.h>

// ---------------------------------------------------------------------------
// D-MPNN (Chemprop-style) forward for MI455X (gfx1250), bf16 WMMA GEMMs.
// ---------------------------------------------------------------------------

typedef __attribute__((ext_vector_type(16))) __bf16 v16bf;
typedef __attribute__((ext_vector_type(8)))  __bf16 v8bf;
typedef __attribute__((ext_vector_type(4)))  __bf16 v4bf;
typedef __attribute__((ext_vector_type(8)))  float  v8f;

#define NN    100000
#define EE    400000
#define HH    256
#define ATOMF 133
#define BONDF 14
#define GG    512
#define DEPTH 4

// output slice offsets (floats)
#define O_PN 0L
#define O_PE 13300000L
#define O_PG 18900000L
#define O_GE 18900512L
#define O_NR 19031584L
#define O_H  44631584L

#define LSTR 40   // LDS row stride in bf16 (80B: 16B aligned, bank-spread)

enum { MODE_PLAIN = 0, MODE_INIT = 1, MODE_MSG = 2, MODE_NODE = 3 };

struct GemmArgs {
  const float* X;      // plain source [rows, inDim]
  const float* atom;   // atom feats [N,133]
  const float* edgef;  // edge feats [E,14]
  const float* Mnode;  // scatter result [N,256] (also node_agg)
  const float* hprev;  // previous edge hidden [E,256]
  const float* ew;     // edge weights [E]
  const int*   src;    // edge_src [E]
  const __bf16* Wb;    // padded bf16 weights [64*TPW, kPad]
  const float* bias;   // [outDim] or nullptr
  float* Y;            // [rows, outDim]
  int rows, inDim, kPad, outDim, relu;
};

// --------------------------- helper kernels --------------------------------

__global__ void k_cvt_w(const float* __restrict__ W, __bf16* __restrict__ Wb,
                        int outDim, int inDim, int outPad, int kPad) {
  int idx = blockIdx.x * blockDim.x + threadIdx.x;
  if (idx >= outPad * kPad) return;
  int o = idx / kPad, k = idx - o * kPad;
  float v = (o < outDim && k < inDim) ? W[(long)o * inDim + k] : 0.0f;
  Wb[idx] = (__bf16)v;
}

__global__ void k_zero(float* __restrict__ p, long n) {
  long i = (long)blockIdx.x * blockDim.x + threadIdx.x;
  long stride = (long)gridDim.x * blockDim.x;
  for (; i < n; i += stride) p[i] = 0.0f;
}

// M[dst[e], :] += w[e] * h[e, :]   (4 floats per thread)
__global__ void k_scatter(const float* __restrict__ h, const float* __restrict__ w,
                          const int* __restrict__ dst, float* __restrict__ M, int Erows) {
  long idx = (long)blockIdx.x * blockDim.x + threadIdx.x;
  if (idx >= (long)Erows * 64) return;
  int e  = (int)(idx >> 6);
  int kq = (int)(idx & 63) << 2;
  const float4 hv = *(const float4*)(h + (long)e * HH + kq);
  float we = w[e];
  float* base = M + (long)dst[e] * HH + kq;
  unsafeAtomicAdd(base + 0, we * hv.x);
  unsafeAtomicAdd(base + 1, we * hv.y);
  unsafeAtomicAdd(base + 2, we * hv.z);
  unsafeAtomicAdd(base + 3, we * hv.w);
}

// out[g[n], :] += x[n, :]
__global__ void k_gscatter(const float* __restrict__ x, const int* __restrict__ g,
                           float* __restrict__ out, int rows) {
  long idx = (long)blockIdx.x * blockDim.x + threadIdx.x;
  if (idx >= (long)rows * 64) return;
  int n  = (int)(idx >> 6);
  int kq = (int)(idx & 63) << 2;
  const float4 v = *(const float4*)(x + (long)n * HH + kq);
  float* base = out + (long)g[n] * HH + kq;
  unsafeAtomicAdd(base + 0, v.x);
  unsafeAtomicAdd(base + 1, v.y);
  unsafeAtomicAdd(base + 2, v.z);
  unsafeAtomicAdd(base + 3, v.w);
}

// pred_graph[g] = dot(tg[g,:], w[0,:]) + b[0]
__global__ void k_graph_head(const float* __restrict__ tg, const float* __restrict__ w,
                             const float* __restrict__ b, float* __restrict__ out) {
  int g = blockIdx.x * blockDim.x + threadIdx.x;
  if (g >= GG) return;
  float s = 0.0f;
  #pragma unroll 8
  for (int k = 0; k < HH; ++k) s += tg[(long)g * HH + k] * w[k];
  out[g] = s + b[0];
}

// --------------------------- WMMA GEMM kernel ------------------------------
// Block = 128 threads (4 wave32), computes a 32-row output stripe (2 row-
// tiles). Wave w owns TPW contiguous 16-col tiles (outPad = 64*TPW).
// A tile staged into LDS as *bf16* (f32->bf16 converted once, cooperatively);
// each A fragment is then two contiguous 16B ds loads per lane (contiguous K
// order: lane holds K = half*16 .. half*16+15, same as B).
// B fragment = one contiguous 32B bf16 load per lane, reused for 2 row-tiles.
// All acc / fragment indices are compile-time constants (no v_movrel).

template <int MODE, int TPW>
__global__ __launch_bounds__(128) void k_gemm(GemmArgs a) {
  __shared__ __bf16 ldsb[32 * LSTR];
  const int tid    = threadIdx.x;
  const int wave   = tid >> 5;
  const int lane   = tid & 31;
  const int lane15 = lane & 15;
  const int half   = lane >> 4;
  const long row0  = (long)blockIdx.x * 32;
  const int nkt    = a.kPad >> 5;

  v8f acc[2][TPW] = {};

  for (int kt = 0; kt < nkt; ++kt) {
    // ---- stage bf16 A tile [32 x 32] into LDS ----
    if constexpr (MODE == MODE_PLAIN || MODE == MODE_MSG) {
      // inDim == kPad == 256 for these modes: aligned float4 path
      #pragma unroll
      for (int j = 0; j < 2; ++j) {
        int q = j * 128 + tid;        // float4 chunk id, 0..255
        int r = q >> 3;
        int c = (q & 7) << 2;         // k offset within 32-wide tile
        long rowg = row0 + r;
        float4 v = make_float4(0.f, 0.f, 0.f, 0.f);
        if (rowg < a.rows) {
          if constexpr (MODE == MODE_PLAIN) {
            v = *(const float4*)(a.X + rowg * a.inDim + kt * 32 + c);
          } else {
            long rev = rowg ^ 1;      // b2rev = e ^ 1 per setup, always valid
            const float4 m4 = *(const float4*)(a.Mnode + (long)a.src[rowg] * HH + kt * 32 + c);
            const float4 h4 = *(const float4*)(a.hprev + rev * HH + kt * 32 + c);
            float w = a.ew[rev];
            v = make_float4(m4.x - w * h4.x, m4.y - w * h4.y,
                            m4.z - w * h4.z, m4.w - w * h4.w);
          }
        }
        v4bf b;
        b[0] = (__bf16)v.x; b[1] = (__bf16)v.y;
        b[2] = (__bf16)v.z; b[3] = (__bf16)v.w;
        *(v4bf*)&ldsb[r * LSTR + c] = b;   // 8B-aligned ds_store_b64
      }
    } else {
      // irregular concat modes: scalar path with boundary guards
      #pragma unroll
      for (int j = 0; j < 8; ++j) {
        int e  = j * 128 + tid;       // 0..1023, coalesced spans
        int r  = e >> 5;
        int kk = e & 31;
        int k  = kt * 32 + kk;
        long rowg = row0 + r;
        float v = 0.0f;
        if (rowg < a.rows) {
          if constexpr (MODE == MODE_INIT) {
            if (k < ATOMF)              v = a.atom[(long)a.src[rowg] * ATOMF + k];
            else if (k < ATOMF + BONDF) v = a.edgef[rowg * BONDF + (k - ATOMF)];
          } else { // MODE_NODE
            if (k < ATOMF)           v = a.atom[rowg * ATOMF + k];
            else if (k < ATOMF + HH) v = a.Mnode[rowg * HH + (k - ATOMF)];
          }
        }
        ldsb[r * LSTR + kk] = (__bf16)v;
      }
    }
    __syncthreads();

    // ---- A fragments: two contiguous 16B LDS loads per row-tile ----
    v16bf af[2];
    #pragma unroll
    for (int rt = 0; rt < 2; ++rt) {
      const int rowA = rt * 16 + lane15;
      const v8bf lo = *(const v8bf*)&ldsb[rowA * LSTR + half * 16];
      const v8bf hi = *(const v8bf*)&ldsb[rowA * LSTR + half * 16 + 8];
      af[rt] = __builtin_shufflevector(lo, hi, 0, 1, 2, 3, 4, 5, 6, 7,
                                       8, 9, 10, 11, 12, 13, 14, 15);
    }

    // ---- per column tile: load B frag once, feed 2 WMMAs ----
    #pragma unroll
    for (int li = 0; li < TPW; ++li) {
      int n = (wave * TPW + li) * 16 + lane15;
      const __bf16* bp = a.Wb + (long)n * a.kPad + kt * 32 + half * 16;
      v16bf bf = *(const v16bf*)bp;  // 32B aligned (kPad*2 % 32 == 0)
      acc[0][li] = __builtin_amdgcn_wmma_f32_16x16x32_bf16(
          false, af[0], false, bf, (short)0, acc[0][li], false, false);
      acc[1][li] = __builtin_amdgcn_wmma_f32_16x16x32_bf16(
          false, af[1], false, bf, (short)0, acc[1][li], false, false);
    }
    __syncthreads();
  }

  // ---- epilogue: bias + relu + guarded stores ----
  #pragma unroll
  for (int li = 0; li < TPW; ++li) {
    int n = (wave * TPW + li) * 16 + lane15;
    if (n < a.outDim) {
      float bv = a.bias ? a.bias[n] : 0.0f;
      #pragma unroll
      for (int rt = 0; rt < 2; ++rt) {
        #pragma unroll
        for (int r = 0; r < 8; ++r) {
          long rowg = row0 + rt * 16 + r + half * 8;
          if (rowg < a.rows) {
            float v = acc[rt][li][r] + bv;
            if (a.relu) v = fmaxf(v, 0.0f);
            a.Y[rowg * a.outDim + n] = v;
          }
        }
      }
    }
  }
}

// ------------------------------- launcher ----------------------------------

extern "C" void kernel_launch(void* const* d_in, const int* in_sizes, int n_in,
                              void* d_out, int out_size, void* d_ws, size_t ws_size,
                              hipStream_t stream) {
  (void)in_sizes; (void)n_in; (void)out_size; (void)ws_size;
  const float* atom  = (const float*)d_in[0];
  const float* edgef = (const float*)d_in[1];
  const float* ew    = (const float*)d_in[2];
  const float* Wi    = (const float*)d_in[3];
  const float* Wm_w  = (const float*)d_in[4];  const float* Wm_b  = (const float*)d_in[5];
  const float* Wn_w  = (const float*)d_in[6];  const float* Wn_b  = (const float*)d_in[7];
  const float* nh0_w = (const float*)d_in[8];  const float* nh0_b = (const float*)d_in[9];
  const float* nh2_w = (const float*)d_in[10]; const float* nh2_b = (const float*)d_in[11];
  const float* eh0_w = (const float*)d_in[12]; const float* eh0_b = (const float*)d_in[13];
  const float* eh2_w = (const float*)d_in[14]; const float* eh2_b = (const float*)d_in[15];
  const float* gh0_w = (const float*)d_in[16]; const float* gh0_b = (const float*)d_in[17];
  const float* gh2_w = (const float*)d_in[18]; const float* gh2_b = (const float*)d_in[19];
  const int* esrc = (const int*)d_in[20];
  const int* edst = (const int*)d_in[21];
  const int* n2g  = (const int*)d_in[23];
  float* out = (float*)d_out;

  // ---- workspace carve (256B aligned) ----
  char* p = (char*)d_ws;
  auto alloc = [&](size_t bytes) -> char* {
    char* r = p; p += (bytes + 255) & ~(size_t)255; return r;
  };
  __bf16* WiB = (__bf16*)alloc((size_t)256 * 160 * 2);
  __bf16* WmB = (__bf16*)alloc((size_t)256 * 256 * 2);
  __bf16* WnB = (__bf16*)alloc((size_t)256 * 416 * 2);
  __bf16* n0B = (__bf16*)alloc((size_t)256 * 256 * 2);
  __bf16* n2B = (__bf16*)alloc((size_t)192 * 256 * 2);   // 133 -> pad 192 (TPW=3)
  __bf16* e0B = (__bf16*)alloc((size_t)256 * 256 * 2);
  __bf16* e2B = (__bf16*)alloc((size_t)64  * 256 * 2);   // 14  -> pad 64  (TPW=1)
  __bf16* g0B = (__bf16*)alloc((size_t)256 * 256 * 2);
  float* hA = (float*)alloc((size_t)EE * HH * 4);
  float* hB = (float*)alloc((size_t)EE * HH * 4);
  float* Mb = (float*)alloc((size_t)NN * HH * 4);

  auto cvt = [&](const float* W, __bf16* Wb, int o, int i, int op, int kp) {
    int tot = op * kp;
    k_cvt_w<<<(tot + 255) / 256, 256, 0, stream>>>(W, Wb, o, i, op, kp);
  };
  cvt(Wi,    WiB, 256, 147, 256, 160);
  cvt(Wm_w,  WmB, 256, 256, 256, 256);
  cvt(Wn_w,  WnB, 256, 389, 256, 416);
  cvt(nh0_w, n0B, 256, 256, 256, 256);
  cvt(nh2_w, n2B, 133, 256, 192, 256);
  cvt(eh0_w, e0B, 256, 256, 256, 256);
  cvt(eh2_w, e2B, 14,  256, 64,  256);
  cvt(gh0_w, g0B, 256, 256, 256, 256);

  GemmArgs g{};
  g.atom = atom; g.edgef = edgef; g.ew = ew; g.src = esrc;

  // ---- initial edge hidden: h0 = relu([atom[src], edgef] @ Wi^T) ----
  g.Wb = WiB; g.bias = nullptr; g.Y = hA;
  g.rows = EE; g.inDim = 147; g.kPad = 160; g.outDim = HH; g.relu = 1;
  k_gemm<MODE_INIT, 4><<<EE / 32, 128, 0, stream>>>(g);

  // ---- depth loop ----
  float* outH = out + O_H;
  float* cur = hA; float* oth = hB;
  for (int d = 0; d < DEPTH; ++d) {
    k_zero<<<4096, 256, 0, stream>>>(Mb, (long)NN * HH);
    k_scatter<<<(int)(((long)EE * 64) / 256), 256, 0, stream>>>(cur, ew, edst, Mb, EE);
    float* nxt = (d == DEPTH - 1) ? outH : oth;
    g.Mnode = Mb; g.hprev = cur; g.Wb = WmB; g.bias = Wm_b; g.Y = nxt;
    g.rows = EE; g.inDim = HH; g.kPad = HH; g.outDim = HH; g.relu = 1;
    k_gemm<MODE_MSG, 4><<<EE / 32, 128, 0, stream>>>(g);
    oth = cur; cur = nxt;
  }

  // ---- node_agg + node_repr ----
  k_zero<<<4096, 256, 0, stream>>>(Mb, (long)NN * HH);
  k_scatter<<<(int)(((long)EE * 64) / 256), 256, 0, stream>>>(outH, ew, edst, Mb, EE);
  g.Mnode = Mb; g.Wb = WnB; g.bias = Wn_b; g.Y = out + O_NR;
  g.rows = NN; g.inDim = 389; g.kPad = 416; g.outDim = HH; g.relu = 1;
  k_gemm<MODE_NODE, 4><<<NN / 32, 128, 0, stream>>>(g);

  // ---- graph embeds (segment sum of node_repr) ----
  k_zero<<<512, 256, 0, stream>>>(out + O_GE, (long)GG * HH);
  k_gscatter<<<(int)(((long)NN * 64) / 256), 256, 0, stream>>>(out + O_NR, n2g, out + O_GE, NN);

  // ---- node head: t = relu(nr @ nh0^T + b); pred_node = t @ nh2^T + b ----
  g.X = out + O_NR; g.Wb = n0B; g.bias = nh0_b; g.Y = hB;
  g.rows = NN; g.inDim = HH; g.kPad = HH; g.outDim = HH; g.relu = 1;
  k_gemm<MODE_PLAIN, 4><<<NN / 32, 128, 0, stream>>>(g);
  g.X = hB; g.Wb = n2B; g.bias = nh2_b; g.Y = out + O_PN;
  g.outDim = ATOMF; g.relu = 0;
  k_gemm<MODE_PLAIN, 3><<<NN / 32, 128, 0, stream>>>(g);

  // ---- edge head ----
  g.X = outH; g.Wb = e0B; g.bias = eh0_b; g.Y = hA;
  g.rows = EE; g.inDim = HH; g.kPad = HH; g.outDim = HH; g.relu = 1;
  k_gemm<MODE_PLAIN, 4><<<EE / 32, 128, 0, stream>>>(g);
  g.X = hA; g.Wb = e2B; g.bias = eh2_b; g.Y = out + O_PE;
  g.outDim = BONDF; g.relu = 0;
  k_gemm<MODE_PLAIN, 1><<<EE / 32, 128, 0, stream>>>(g);

  // ---- graph head ----
  g.X = out + O_GE; g.Wb = g0B; g.bias = gh0_b; g.Y = Mb;
  g.rows = GG; g.inDim = HH; g.kPad = HH; g.outDim = HH; g.relu = 1;
  k_gemm<MODE_PLAIN, 4><<<GG / 32, 128, 0, stream>>>(g);
  k_graph_head<<<2, 256, 0, stream>>>(Mb, gh2_w, gh2_b, out + O_PG);
}
